// Actor_85323820302541
// MI455X (gfx1250) — matrix operand order repaired and checked by reading, not compiled, not run on hardware
//
#include <hip/hip_runtime.h>
#include <hip/hip_bf16.h>
#include <math.h>

// ---------------------------------------------------------------- types
typedef __bf16 bf16_t;
typedef __bf16 v16bf __attribute__((ext_vector_type(16)));
typedef __bf16 v8bf  __attribute__((ext_vector_type(8)));
typedef float  v8f   __attribute__((ext_vector_type(8)));
typedef unsigned int v4u __attribute__((ext_vector_type(4)));

#define BATCH 4096
#define OUTW  4368   // per-row output width: 3+5+264+2048+2048
#define MU_OFF 272
#define LV_OFF 2320

__device__ __forceinline__ float geluf(float x) {
    return 0.5f * x * (1.0f + erff(x * 0.70710678118654752f));
}
__device__ __forceinline__ float sigmf(float x) { return 1.0f / (1.0f + __expf(-x)); }
__device__ __forceinline__ float lreluf(float x) { return x > 0.0f ? x : 0.01f * x; }

// ---------------------------------------------------------------- conversions
__global__ void cvt_f32_bf16(const float* __restrict__ s, bf16_t* __restrict__ d, size_t n) {
    size_t i = (size_t)blockIdx.x * blockDim.x + threadIdx.x;
    size_t stride = (size_t)gridDim.x * blockDim.x;
    for (; i < n; i += stride) d[i] = (bf16_t)s[i];
}

// pad columns cin -> cout with zeros (for fc1: 528 -> 544)
__global__ void cvt_pad_bf16(const float* __restrict__ s, bf16_t* __restrict__ d,
                             int rows, int cin, int cout) {
    size_t n = (size_t)rows * cout;
    size_t i = (size_t)blockIdx.x * blockDim.x + threadIdx.x;
    size_t stride = (size_t)gridDim.x * blockDim.x;
    for (; i < n; i += stride) {
        int r = (int)(i / cout), c = (int)(i % cout);
        d[i] = (c < cin) ? (bf16_t)s[(size_t)r * cin + c] : (bf16_t)0.0f;
    }
}

// ---------------------------------------------------------------- feature prep
// h0 row = [stateVec(264), thickness(132), angles(132), zero-pad(16)]  (bf16, ld=544)
__global__ __launch_bounds__(256) void prep_kernel(const float* __restrict__ st,
                                                   bf16_t* __restrict__ h0) {
    int row = blockIdx.x;
    const float* x = st + (size_t)row * 264;   // (132,2) flattened
    bf16_t* h = h0 + (size_t)row * 544;
    int t = threadIdx.x;
    for (int i = t; i < 264; i += 256) h[i] = (bf16_t)x[i];
    for (int i = t; i < 66; i += 256) {
        h[264 + 2 * i]     = (bf16_t)x[2 * i];                                // x[:,i,0]
        h[264 + 2 * i + 1] = (bf16_t)fabsf(x[2 * i + 1] - x[2 * (66 + i) + 1]);
    }
    for (int i = t; i < 132; i += 256) {
        auto vecx = [&](int k, float& vx, float& vy) {
            int k2 = (k < 131) ? (k + 1) : 0;
            vx = x[2 * k2]     - x[2 * k];
            vy = x[2 * k2 + 1] - x[2 * k + 1];
        };
        float ax, ay, bx, by;
        vecx(i, ax, ay);
        int j = ((i + 1) % 66) * 2;
        vecx(j, bx, by);
        float dot = ax * bx + ay * by;
        float nn  = sqrtf((ax * ax + ay * ay) * (bx * bx + by * by));
        float c   = dot / nn;
        c = fminf(1.0f, fmaxf(-1.0f, c));
        h[396 + i] = (bf16_t)acosf(c);
    }
    for (int i = 528 + t; i < 544; i += 256) h[i] = (bf16_t)0.0f;
}

// ---------------------------------------------------------------- WMMA GEMM
// C[M,N] = act(A[M,K] @ W[N,K]^T + bias) ; A,W bf16 row-major, K % 32 == 0, M % 128 == 0.
// Block tile 128x128, 8 waves, wave tile 32(M) x 64(N) = 2x4 x v_wmma_f32_16x16x32_bf16.
// LDS ping-pong double buffer: next tile's global loads overlap current tile's WMMAs,
// one barrier per K-iteration.
#define EPI_NONE 0
#define EPI_GELU 1
#define EPI_SIG  2

__global__ __launch_bounds__(256) void gemm_bf16_wmma(
    const bf16_t* __restrict__ A, const bf16_t* __restrict__ W,
    const float* __restrict__ bias,
    float* __restrict__ Cf, int ldc,       // optional f32 output (may be strided into d_out)
    bf16_t* __restrict__ Cb, int ldcb,     // optional bf16 output
    int M, int N, int K, int epi)
{
    __shared__ __align__(16) bf16_t As[2][128 * 32];   // [m][k]  8KB each
    __shared__ __align__(16) bf16_t Bs[2][128 * 32];   // [n][k]  8KB each

    const int tid  = threadIdx.x;
    const int wave = tid >> 5;
    const int lane = tid & 31;
    const int l16   = lane & 15;
    const int lhalf = lane >> 4;

    const int block_n = blockIdx.x * 128;
    const int block_m = blockIdx.y * 128;
    const int wm = (wave >> 1) * 32;     // 0,32,64,96
    const int wn = (wave & 1) * 64;      // 0,64

    // tile staging: 128 rows x 32 cols, 2 threads/row, 16 bf16 (2 x b128) each
    const int t_row = tid >> 1;
    const int t_col = (tid & 1) * 16;
    const int s_off = t_row * 32 + t_col;
    const bool bvalid = (block_n + t_row) < N;   // loop-invariant B-row validity

    // zero OOB B rows ONCE (both buffers; never overwritten afterwards)
    if (!bvalid) {
        v4u zv = {0u, 0u, 0u, 0u};
        *(v4u*)&Bs[0][s_off] = zv; *(v4u*)&Bs[0][s_off + 8] = zv;
        *(v4u*)&Bs[1][s_off] = zv; *(v4u*)&Bs[1][s_off + 8] = zv;
    }

    const bf16_t* aG = A + (size_t)(block_m + t_row) * K + t_col;
    const bf16_t* wG = W + (size_t)(block_n + (bvalid ? t_row : 0)) * K + t_col;

    // ---- preload tile 0 into buffer 0
    v4u a0 = *(const v4u*)(aG);
    v4u a1 = *(const v4u*)(aG + 8);
    v4u b0 = {0u,0u,0u,0u}, b1 = {0u,0u,0u,0u};
    if (bvalid) { b0 = *(const v4u*)(wG); b1 = *(const v4u*)(wG + 8); }
    *(v4u*)&As[0][s_off] = a0; *(v4u*)&As[0][s_off + 8] = a1;
    if (bvalid) { *(v4u*)&Bs[0][s_off] = b0; *(v4u*)&Bs[0][s_off + 8] = b1; }
    __syncthreads();

    v8f acc[2][4] = {};
    int buf = 0;
    for (int k0 = 0; k0 < K; k0 += 32) {
        const bool hasNext = (k0 + 32) < K;
        // ---- issue next tile's global loads early (overlap with WMMA below)
        if (hasNext) {
            const bf16_t* an = aG + k0 + 32;
            a0 = *(const v4u*)(an); a1 = *(const v4u*)(an + 8);
            if (bvalid) {
                const bf16_t* wn = wG + k0 + 32;
                b0 = *(const v4u*)(wn); b1 = *(const v4u*)(wn + 8);
            }
            if (k0 + 64 < K) {
                __builtin_prefetch(aG + k0 + 64, 0, 1);
                __builtin_prefetch(wG + k0 + 64, 0, 1);
            }
        }

        // ---- fragments from current buffer
        const bf16_t* __restrict__ Ab = &As[buf][0];
        const bf16_t* __restrict__ Bb = &Bs[buf][0];
        // A: lane holds row m, K elems {kb..kb+7} U {kb+16..kb+23}, kb = 8*lhalf
        v16bf af[2];
        #pragma unroll
        for (int i = 0; i < 2; ++i) {
            int m = wm + i * 16 + l16;
            int kb = lhalf * 8;
            v8bf lo = *(const v8bf*)&Ab[m * 32 + kb];
            v8bf hi = *(const v8bf*)&Ab[m * 32 + kb + 16];
            #pragma unroll
            for (int e = 0; e < 8; ++e) { af[i][e] = lo[e]; af[i][8 + e] = hi[e]; }
        }
        // B: lane holds col n, K elems {kb..kb+7} U {kb+16..kb+23}, kb = 8*lhalf
        v16bf bfr[4];
        #pragma unroll
        for (int j = 0; j < 4; ++j) {
            int n2 = wn + j * 16 + l16;
            int kb = lhalf * 8;
            v8bf lo = *(const v8bf*)&Bb[n2 * 32 + kb];
            v8bf hi = *(const v8bf*)&Bb[n2 * 32 + kb + 16];
            #pragma unroll
            for (int e = 0; e < 8; ++e) { bfr[j][e] = lo[e]; bfr[j][8 + e] = hi[e]; }
        }
        #pragma unroll
        for (int i = 0; i < 2; ++i)
            #pragma unroll
            for (int j = 0; j < 4; ++j)
                acc[i][j] = __builtin_amdgcn_wmma_f32_16x16x32_bf16(
                    false, af[i], false, bfr[j], (short)0, acc[i][j], false, false);

        // ---- stash next tile into the other buffer (no conflict with reads above)
        if (hasNext) {
            int nb = buf ^ 1;
            *(v4u*)&As[nb][s_off] = a0; *(v4u*)&As[nb][s_off + 8] = a1;
            if (bvalid) { *(v4u*)&Bs[nb][s_off] = b0; *(v4u*)&Bs[nb][s_off + 8] = b1; }
        }
        __syncthreads();           // single barrier per iteration
        buf ^= 1;
    }

    // ---- epilogue: VGPR r -> M = base + r + 8*lhalf ; lane l16 -> N
    #pragma unroll
    for (int i = 0; i < 2; ++i) {
        #pragma unroll
        for (int j = 0; j < 4; ++j) {
            int n = block_n + wn + j * 16 + l16;
            if (n >= N) continue;
            float bv = bias ? bias[n] : 0.0f;
            #pragma unroll
            for (int r = 0; r < 8; ++r) {
                int m = block_m + wm + i * 16 + r + 8 * lhalf;
                float v = acc[i][j][r] + bv;
                if (epi == EPI_GELU)     v = geluf(v);
                else if (epi == EPI_SIG) v = sigmf(v);
                if (Cf) Cf[(size_t)m * ldc + n] = v;
                if (Cb) Cb[(size_t)m * ldcb + n] = (bf16_t)v;
            }
        }
    }
}

// ---------------------------------------------------------------- row reductions
__device__ __forceinline__ void blockMeanInv(const float* __restrict__ x, int width,
                                             int tid, float& mean, float& inv) {
    __shared__ float rs[8], rs2[8];
    float s = 0.0f, s2 = 0.0f;
    for (int c = tid; c < width; c += 256) { float v = x[c]; s += v; s2 += v * v; }
    #pragma unroll
    for (int o = 16; o > 0; o >>= 1) { s += __shfl_down(s, o, 32); s2 += __shfl_down(s2, o, 32); }
    if ((tid & 31) == 0) { rs[tid >> 5] = s; rs2[tid >> 5] = s2; }
    __syncthreads();
    if (tid == 0) {
        float S = 0.0f, S2 = 0.0f;
        for (int i = 0; i < 8; ++i) { S += rs[i]; S2 += rs2[i]; }
        rs[0] = S; rs2[0] = S2;
    }
    __syncthreads();
    mean = rs[0] / width;
    float var = rs2[0] / width - mean * mean;
    inv = rsqrtf(var + 1e-5f);
}

// out = lrelu(LN(in)*g + b) as bf16, one row per block
__global__ __launch_bounds__(256) void ln_lrelu_kernel(const float* __restrict__ in, int width,
                                                       const float* __restrict__ g,
                                                       const float* __restrict__ b,
                                                       bf16_t* __restrict__ out) {
    int row = blockIdx.x;
    const float* x = in + (size_t)row * width;
    float mean, inv; blockMeanInv(x, width, threadIdx.x, mean, inv);
    for (int c = threadIdx.x; c < width; c += 256) {
        float v = (x[c] - mean) * inv * g[c] + b[c];
        out[(size_t)row * width + c] = (bf16_t)lreluf(v);
    }
}

// z = mu + eps*exp(0.5*logvar), mu/logvar read strided from d_out
__global__ void z_kernel(const float* __restrict__ dout, const float* __restrict__ eps,
                         bf16_t* __restrict__ z) {
    size_t i = (size_t)blockIdx.x * blockDim.x + threadIdx.x;
    if (i >= (size_t)BATCH * 2048) return;
    size_t r = i >> 11; int c = (int)(i & 2047);
    float mu = dout[r * OUTW + MU_OFF + c];
    float lv = dout[r * OUTW + LV_OFF + c];
    z[i] = (bf16_t)(mu + eps[i] * __expf(0.5f * lv));
}

// r1 tail: x0 = gelu(lrelu(LN(t2)*g+b + logvar)); x0pl = bf16(x0 + logvar)
__global__ __launch_bounds__(256) void r1_tail_kernel(const float* __restrict__ xr,
                                                      const float* __restrict__ g,
                                                      const float* __restrict__ bb,
                                                      const float* __restrict__ dout,
                                                      float* __restrict__ x0f,
                                                      bf16_t* __restrict__ x0pl) {
    int row = blockIdx.x;
    const float* x  = xr + (size_t)row * 2048;
    const float* lv = dout + (size_t)row * OUTW + LV_OFF;
    float mean, inv; blockMeanInv(x, 2048, threadIdx.x, mean, inv);
    for (int c = threadIdx.x; c < 2048; c += 256) {
        float l = lv[c];
        float y = (x[c] - mean) * inv * g[c] + bb[c] + l;
        float x0 = geluf(lreluf(y));
        x0f[(size_t)row * 2048 + c] = x0;
        x0pl[(size_t)row * 2048 + c] = (bf16_t)(x0 + l);
    }
}

// r2 tail: x1 = gelu(lrelu(LN(t)*g+b + (x0+logvar))); s = x1+x0; x2 = s[:1024]*sigmoid(s[1024:])
__global__ __launch_bounds__(256) void r2_tail_kernel(const float* __restrict__ xr,
                                                      const float* __restrict__ g,
                                                      const float* __restrict__ bb,
                                                      const float* __restrict__ dout,
                                                      const float* __restrict__ x0f,
                                                      bf16_t* __restrict__ x2) {
    int row = blockIdx.x;
    __shared__ float sbuf[2048];
    const float* x  = xr  + (size_t)row * 2048;
    const float* lv = dout + (size_t)row * OUTW + LV_OFF;
    const float* x0 = x0f + (size_t)row * 2048;
    float mean, inv; blockMeanInv(x, 2048, threadIdx.x, mean, inv);
    for (int c = threadIdx.x; c < 2048; c += 256) {
        float y = (x[c] - mean) * inv * g[c] + bb[c] + (x0[c] + lv[c]);
        sbuf[c] = geluf(lreluf(y)) + x0[c];
    }
    __syncthreads();
    for (int c = threadIdx.x; c < 1024; c += 256)
        x2[(size_t)row * 1024 + c] = (bf16_t)(sbuf[c] * sigmf(sbuf[c + 1024]));
}

// ---------------------------------------------------------------- small heads
__global__ __launch_bounds__(128) void heads_kernel(
    const float* __restrict__ x3,
    const float* aw, const float* ab,
    const float* p10w, const float* p10b, const float* p20w, const float* p20b,
    const float* p30w, const float* p30b, const float* p40w, const float* p40b,
    const float* p11w, const float* p11b, const float* p21w, const float* p21b,
    const float* p12w, const float* p12b, const float* p22w, const float* p22b,
    float* __restrict__ out) {
    int row = blockIdx.x, t = threadIdx.x;
    __shared__ float xs[128], h0[16], h1[16], h2[16], lg[3];
    xs[t] = x3[(size_t)row * 128 + t];
    __syncthreads();
    if (t < 3)              { float a = ab[t];      for (int k = 0; k < 128; ++k) a += aw[t * 128 + k] * xs[k];        lg[t] = geluf(a); }
    if (t >= 32 && t < 48)  { int o = t - 32; float a = p10b[o]; for (int k = 0; k < 128; ++k) a += p10w[o * 128 + k] * xs[k]; h0[o] = geluf(a); }
    if (t >= 64 && t < 80)  { int o = t - 64; float a = p11b[o]; for (int k = 0; k < 128; ++k) a += p11w[o * 128 + k] * xs[k]; h1[o] = fmaxf(a, 0.0f); }
    if (t >= 96 && t < 112) { int o = t - 96; float a = p12b[o]; for (int k = 0; k < 128; ++k) a += p12w[o * 128 + k] * xs[k]; h2[o] = fmaxf(a, 0.0f); }
    __syncthreads();
    float* orow = out + (size_t)row * OUTW;
    const float C = 0.002f;
    if (t == 0) {
        float mx = fmaxf(lg[0], fmaxf(lg[1], lg[2]));
        float e0 = expf(lg[0] - mx), e1 = expf(lg[1] - mx), e2 = expf(lg[2] - mx);
        float ss = e0 + e1 + e2;
        orow[0] = e0 / ss; orow[1] = e1 / ss; orow[2] = e2 / ss;
    }
    if (t == 1) { float a = p20b[0]; for (int k = 0; k < 16; ++k) a += p20w[k] * h0[k];
                  float s = sigmf(a); orow[3] = s + 0.1f * s * (1.0f - s); }
    if (t == 2) { float a = p30b[0]; for (int k = 0; k < 16; ++k) a += p30w[k] * h0[k];
                  float th = tanhf(a); orow[4] = C * (th - 0.1f * th * th); }
    if (t == 3) { float a = p40b[0]; for (int k = 0; k < 16; ++k) a += p40w[k] * h0[k];
                  float th = tanhf(a); orow[5] = C * (th - 0.1f * th * th); }
    if (t == 4) { float a = p21b[0]; for (int k = 0; k < 16; ++k) a += p21w[k] * h1[k];
                  float th = tanhf(a); orow[6] = C * (th - 0.1f * th * th); }
    if (t == 5) { float a = p22b[0]; for (int k = 0; k < 16; ++k) a += p22w[k] * h2[k];
                  float th = tanhf(a); orow[7] = C * (th - 0.1f * th * th); }
}

// ---------------------------------------------------------------- launch
enum {
    IN_STATE = 0, IN_EPS,
    FC1_W, FC1_B, FC21_W, FC21_B, FC22_W, FC22_B, FC3_W, FC3_B, FC4_W, FC4_B,
    R1FC1_W, R1FC1_B, R1FC2_W, R1FC2_B, R1LN1_G, R1LN1_B, R1LN2_G, R1LN2_B,
    R2FC1_W, R2FC1_B, R2FC2_W, R2FC2_B, R2LN1_G, R2LN1_B, R2LN2_G, R2LN2_B,
    FC_W, FC_B, ACT_W, ACT_B,
    P10_W, P10_B, P20_W, P20_B, P30_W, P30_B, P40_W, P40_B,
    P11_W, P11_B, P21_W, P21_B,
    P12_W, P12_B, P22_W, P22_B
};

extern "C" void kernel_launch(void* const* d_in, const int* in_sizes, int n_in,
                              void* d_out, int out_size, void* d_ws, size_t ws_size,
                              hipStream_t stream) {
    (void)in_sizes; (void)n_in; (void)out_size; (void)ws_size;
    const float* F[48];
    for (int i = 0; i < 48; ++i) F[i] = (const float*)d_in[i];
    float* out = (float*)d_out;

    // -------- workspace layout
    char* ws = (char*)d_ws;
    size_t off = 0;
    auto ab = [&](size_t elems, size_t esz) { size_t o = off; off = (off + elems * esz + 255) & ~(size_t)255; return o; };
    auto B = [&](size_t elems) { return (bf16_t*)(ws + ab(elems, 2)); };
    auto Fp = [&](size_t elems) { return (float*)(ws + ab(elems, 4)); };

    bf16_t* w_fc1   = B((size_t)1024 * 544);
    bf16_t* w_fc21  = B((size_t)2048 * 1024);
    bf16_t* w_fc22  = B((size_t)2048 * 1024);
    bf16_t* w_fc3   = B((size_t)1024 * 2048);
    bf16_t* w_fc4   = B((size_t)264 * 1024);
    bf16_t* w_r1a   = B((size_t)8192 * 2048);
    bf16_t* w_r1b   = B((size_t)2048 * 8192);
    bf16_t* w_r2a   = B((size_t)4096 * 2048);
    bf16_t* w_r2b   = B((size_t)2048 * 4096);
    bf16_t* w_fc    = B((size_t)128 * 1024);
    bf16_t* h0_bf   = B((size_t)BATCH * 544);
    bf16_t* h1_bf   = B((size_t)BATCH * 1024);   // reused as x2_bf
    bf16_t* lv_bf   = B((size_t)BATCH * 2048);
    bf16_t* z_bf    = B((size_t)BATCH * 2048);   // reused as (x0+logvar)_bf
    bf16_t* h3_bf   = B((size_t)BATCH * 1024);
    bf16_t* big_bf  = B((size_t)BATCH * 8192);   // LN outputs
    float*  big_f   = Fp((size_t)BATCH * 8192);  // GEMM f32 outputs pre-LN
    float*  xr_f    = Fp((size_t)BATCH * 2048);
    float*  x0_f    = Fp((size_t)BATCH * 2048);
    float*  x3_f    = Fp((size_t)BATCH * 128);

    // -------- weight conversion (every call: deterministic, no caching)
    auto cvt = [&](int idx, bf16_t* dst, size_t n) {
        int blocks = (int)((n + 255) / 256); if (blocks > 32768) blocks = 32768;
        cvt_f32_bf16<<<blocks, 256, 0, stream>>>(F[idx], dst, n);
    };
    cvt_pad_bf16<<<2048, 256, 0, stream>>>(F[FC1_W], w_fc1, 1024, 528, 544);
    cvt(FC21_W, w_fc21, (size_t)2048 * 1024);
    cvt(FC22_W, w_fc22, (size_t)2048 * 1024);
    cvt(FC3_W,  w_fc3,  (size_t)1024 * 2048);
    cvt(FC4_W,  w_fc4,  (size_t)264 * 1024);
    cvt(R1FC1_W, w_r1a, (size_t)8192 * 2048);
    cvt(R1FC2_W, w_r1b, (size_t)2048 * 8192);
    cvt(R2FC1_W, w_r2a, (size_t)4096 * 2048);
    cvt(R2FC2_W, w_r2b, (size_t)2048 * 4096);
    cvt(FC_W,   w_fc,   (size_t)128 * 1024);

    // -------- features
    prep_kernel<<<BATCH, 256, 0, stream>>>(F[IN_STATE], h0_bf);

    auto gemm = [&](const bf16_t* A, const bf16_t* W, const float* bias,
                    float* Cf, int ldc, bf16_t* Cb, int ldcb, int N, int K, int epi) {
        dim3 g((N + 127) / 128, BATCH / 128);
        gemm_bf16_wmma<<<g, 256, 0, stream>>>(A, W, bias, Cf, ldc, Cb, ldcb, BATCH, N, K, epi);
    };

    // encoder
    gemm(h0_bf, w_fc1,  F[FC1_B],  nullptr, 0, h1_bf, 1024, 1024, 544,  EPI_GELU);      // h1
    gemm(h1_bf, w_fc21, F[FC21_B], out + MU_OFF, OUTW, nullptr, 0, 2048, 1024, EPI_NONE); // mu -> d_out
    gemm(h1_bf, w_fc22, F[FC22_B], out + LV_OFF, OUTW, lv_bf, 2048, 2048, 1024, EPI_NONE);// logvar -> d_out + bf16
    z_kernel<<<(BATCH * 2048 + 255) / 256, 256, 0, stream>>>(out, F[IN_EPS], z_bf);
    gemm(z_bf,  w_fc3,  F[FC3_B],  nullptr, 0, h3_bf, 1024, 1024, 2048, EPI_GELU);      // gelu(fc3)
    gemm(h3_bf, w_fc4,  F[FC4_B],  out + 8, OUTW, nullptr, 0, 264, 1024, EPI_SIG);      // recon -> d_out

    // resblock 1 (width 8192)
    gemm(lv_bf, w_r1a, F[R1FC1_B], big_f, 8192, nullptr, 0, 8192, 2048, EPI_NONE);
    ln_lrelu_kernel<<<BATCH, 256, 0, stream>>>(big_f, 8192, F[R1LN1_G], F[R1LN1_B], big_bf);
    gemm(big_bf, w_r1b, F[R1FC2_B], xr_f, 2048, nullptr, 0, 2048, 8192, EPI_NONE);
    r1_tail_kernel<<<BATCH, 256, 0, stream>>>(xr_f, F[R1LN2_G], F[R1LN2_B], out, x0_f, z_bf);

    // resblock 2 (width 4096)
    gemm(z_bf, w_r2a, F[R2FC1_B], big_f, 4096, nullptr, 0, 4096, 2048, EPI_NONE);
    ln_lrelu_kernel<<<BATCH, 256, 0, stream>>>(big_f, 4096, F[R2LN1_G], F[R2LN1_B], big_bf);
    gemm(big_bf, w_r2b, F[R2FC2_B], xr_f, 2048, nullptr, 0, 2048, 4096, EPI_NONE);
    r2_tail_kernel<<<BATCH, 256, 0, stream>>>(xr_f, F[R2LN2_G], F[R2LN2_B], out, x0_f, h1_bf);

    // trunk + heads
    gemm(h1_bf, w_fc, F[FC_B], x3_f, 128, nullptr, 0, 128, 1024, EPI_GELU);
    heads_kernel<<<BATCH, 128, 0, stream>>>(x3_f,
        F[ACT_W], F[ACT_B],
        F[P10_W], F[P10_B], F[P20_W], F[P20_B], F[P30_W], F[P30_B], F[P40_W], F[P40_B],
        F[P11_W], F[P11_B], F[P21_W], F[P21_B],
        F[P12_W], F[P12_B], F[P22_W], F[P22_B],
        out);
}
